// TreeGenerator_85409719648826
// MI455X (gfx1250) — matrix operand order repaired and checked by reading, not compiled, not run on hardware
//
#include <hip/hip_runtime.h>
#include <hip/hip_bf16.h>
#include <math.h>

#define NUM_RULES 8192
#define NUM_NT 16
#define EMBD 1024
#define HID 1024
#define MAX_STEPS 1024
#define MAX_STACK (MAX_STEPS + 4)
#define PADV (-1)

typedef __attribute__((ext_vector_type(16))) __bf16 v16bf;
typedef __attribute__((ext_vector_type(8)))  float v8f;

__device__ __forceinline__ unsigned short f2bf(float x) {
  __bf16 h = (__bf16)x;
  return __builtin_bit_cast(unsigned short, h);
}

// ---------------- threefry2x32 (JAX-style PRNG) ----------------
__device__ __forceinline__ void threefry2x32(unsigned k0, unsigned k1,
                                             unsigned x0, unsigned x1,
                                             unsigned* o0, unsigned* o1) {
  const unsigned rots[8] = {13u,15u,26u,6u,17u,29u,16u,24u};
  unsigned ks[3] = {k0, k1, k0 ^ k1 ^ 0x1BD11BDAu};
  x0 += ks[0]; x1 += ks[1];
  #pragma unroll
  for (int r = 0; r < 5; ++r) {
    #pragma unroll
    for (int i = 0; i < 4; ++i) {
      unsigned rr = rots[(r * 4 + i) & 7];
      x0 += x1;
      x1 = (x1 << rr) | (x1 >> (32u - rr));
      x1 ^= x0;
    }
    x0 += ks[(r + 1) % 3];
    x1 += ks[(r + 2) % 3] + (unsigned)(r + 1);
  }
  *o0 = x0; *o1 = x1;
}

__device__ __forceinline__ float gumbel_for(unsigned fk0, unsigned fk1, int r) {
  unsigned o0, o1, bits;
  if (r < NUM_RULES / 2) {
    threefry2x32(fk0, fk1, (unsigned)r, (unsigned)(r + NUM_RULES / 2), &o0, &o1);
    bits = o0;
  } else {
    threefry2x32(fk0, fk1, (unsigned)(r - NUM_RULES / 2), (unsigned)r, &o0, &o1);
    bits = o1;
  }
  float u = __uint_as_float((bits >> 9) | 0x3f800000u) - 1.0f;
  return -logf(-logf(u));
}

// ---------------- wave-level bf16 WMMA matvec ----------------
// Weights fragment-major: frag(rt,kt) = 512 contiguous bf16 (32 lanes x 16
// elems), lane/elem -> (M,K): M=lane%16, K=16*(lane/16)+elem. Vector is pre-converted
// bf16; B column n==0 holds it (lane0: K kb..kb+15, lane16: K kb+16..kb+31).
// Zero-fill of B is hoisted: masked loads leave lanes n!=0 at zero forever.
__device__ __forceinline__ v8f wave_matvec(const unsigned short* __restrict__ frag,
                                           size_t fragBase, int ktBegin, int ktCount,
                                           const unsigned short* __restrict__ vecb,
                                           int lane) {
  v8f acc = {};
  int n  = lane & 15;
  int hi = lane >> 4;
  const unsigned short* aptr = frag + (fragBase + (size_t)ktBegin) * 512 + (size_t)lane * 16;
  const unsigned short* bptr = vecb + ktBegin * 32 + hi * 16;
  v16bf b;
  #pragma unroll
  for (int e = 0; e < 16; ++e) b[e] = (__bf16)0.0f;
  for (int kt = 0; kt < ktCount; ++kt) {
    v16bf a = *(const v16bf*)aptr;
    __builtin_prefetch(aptr + 512, 0, 1);
    aptr += 512;
    if (n == 0) b = *(const v16bf*)bptr;   // only lanes 0/16 overwritten
    bptr += 32;
    acc = __builtin_amdgcn_wmma_f32_16x16x32_bf16(false, a, false, b,
                                                  (short)0, acc, false, false);
  }
  return acc;
}

// ---------------- weight conversion: f32 row-major -> bf16 fragment-major ----
__global__ void k_conv(const float* __restrict__ W, unsigned short* __restrict__ out,
                       int R, int K) {
  int KT = K >> 5;
  size_t total  = (size_t)R * K;
  size_t stride = (size_t)gridDim.x * blockDim.x;
  for (size_t o = (size_t)blockIdx.x * blockDim.x + threadIdx.x; o < total; o += stride) {
    size_t frag = o >> 9;
    int within = (int)(o & 511);
    int lane = within >> 4;
    int e = within & 15;
    size_t rt = frag / KT;
    int kt = (int)(frag % KT);
    size_t M = rt * 16 + (lane & 15);
    int v = e >> 1;
    int k = kt * 32 + ((lane >= 16) ? 16 : 0) + e; (void)v;
    out[o] = f2bf(W[M * K + k]);
  }
}

// ---------------- init ----------------
__global__ void k_init(int* ctl, int* stack, const float* h0, float* hstate,
                       unsigned short* hstateb, const float* emb,
                       unsigned short* xvecb) {
  int tid = threadIdx.x;
  for (int i = tid; i < MAX_STACK * 2; i += blockDim.x) stack[i] = PADV;
  for (int i = tid; i < 2 * HID; i += blockDim.x) {
    float v = h0[i];
    hstate[i] = v;
    hstateb[i] = f2bf(v);
  }
  for (int i = tid; i < EMBD; i += blockDim.x) {  // prev=parent=PAD -> emb row 0
    unsigned short v = f2bf(emb[i]);
    xvecb[i] = v;
    xvecb[EMBD + i] = v;
  }
  if (tid == 0) {
    stack[0] = 0; stack[1] = PADV;
    ctl[0] = 1;    // do_expand (root symbol 0 is NT)
    ctl[1] = 0;    // symbol
    ctl[2] = PADV; // parent action
    ctl[3] = PADV; // prev action
    ctl[4] = 1;    // ptr
    ctl[5] = 0;    // idx
  }
}

// ---------------- GRU layer 1 (x: 2048, h: 1024) ----------------
// 64 blocks x 9 waves. waves 0..5: Wih1 (gate g=w/2, K-half w&1); waves 6..8: Whh1.
__global__ void k_gru1(const int* __restrict__ ctl,
                       const unsigned short* __restrict__ Wih1f,
                       const unsigned short* __restrict__ Whh1f,
                       const float* __restrict__ bih1, const float* __restrict__ bhh1,
                       const unsigned short* __restrict__ xvecb,
                       const float* __restrict__ hstate,
                       const unsigned short* __restrict__ hstateb,
                       float* __restrict__ h1, unsigned short* __restrict__ h1b) {
  if (!ctl[0]) return;
  __shared__ float gi[2][3][16];
  __shared__ float gh[3][16];
  int tid = threadIdx.x, w = tid >> 5, lane = tid & 31, b = blockIdx.x;
  if (w < 6) {
    int g = w >> 1, khalf = w & 1;
    size_t rt = (size_t)g * 64 + b;
    v8f acc = wave_matvec(Wih1f, rt * 64, khalf * 32, 32, xvecb, lane);
    if ((lane & 15) == 0) {
      int half = lane >> 4;
      #pragma unroll
      for (int r = 0; r < 8; ++r) gi[khalf][g][half * 8 + r] = acc[r];
    }
  } else {
    int g = w - 6;
    size_t rt = (size_t)g * 64 + b;
    v8f acc = wave_matvec(Whh1f, rt * 32, 0, 32, hstateb, lane);
    if ((lane & 15) == 0) {
      int half = lane >> 4;
      #pragma unroll
      for (int r = 0; r < 8; ++r) gh[g][half * 8 + r] = acc[r];
    }
  }
  __syncthreads();
  if (tid < 16) {
    int j = b * 16 + tid;
    float ir = gi[0][0][tid] + gi[1][0][tid] + bih1[j];
    float iz = gi[0][1][tid] + gi[1][1][tid] + bih1[HID + j];
    float in_ = gi[0][2][tid] + gi[1][2][tid] + bih1[2 * HID + j];
    float hr = gh[0][tid] + bhh1[j];
    float hz = gh[1][tid] + bhh1[HID + j];
    float hn = gh[2][tid] + bhh1[2 * HID + j];
    float r = 1.0f / (1.0f + expf(-(ir + hr)));
    float z = 1.0f / (1.0f + expf(-(iz + hz)));
    float nn = tanhf(in_ + r * hn);
    float out = (1.0f - z) * nn + z * hstate[j];
    h1[j] = out;
    h1b[j] = f2bf(out);
  }
}

// ---------------- GRU layer 2 (both K=1024) ----------------
// 64 blocks x 6 waves. waves 0..2: Wih2 gates; waves 3..5: Whh2 gates.
__global__ void k_gru2(const int* __restrict__ ctl,
                       const unsigned short* __restrict__ Wih2f,
                       const unsigned short* __restrict__ Whh2f,
                       const float* __restrict__ bih2, const float* __restrict__ bhh2,
                       const unsigned short* __restrict__ h1b,
                       const float* __restrict__ hstate,
                       const unsigned short* __restrict__ hstateb,
                       float* __restrict__ h2, unsigned short* __restrict__ h2b) {
  if (!ctl[0]) return;
  __shared__ float gi[3][16];
  __shared__ float gh[3][16];
  int tid = threadIdx.x, w = tid >> 5, lane = tid & 31, b = blockIdx.x;
  int g = (w < 3) ? w : (w - 3);
  size_t rt = (size_t)g * 64 + b;
  v8f acc = (w < 3) ? wave_matvec(Wih2f, rt * 32, 0, 32, h1b, lane)
                    : wave_matvec(Whh2f, rt * 32, 0, 32, hstateb + HID, lane);
  if ((lane & 15) == 0) {
    int half = lane >> 4;
    float* dst = (w < 3) ? &gi[g][half * 8] : &gh[g][half * 8];
    #pragma unroll
    for (int r = 0; r < 8; ++r) dst[r] = acc[r];
  }
  __syncthreads();
  if (tid < 16) {
    int j = b * 16 + tid;
    float r = 1.0f / (1.0f + expf(-((gi[0][tid] + bih2[j]) + (gh[0][tid] + bhh2[j]))));
    float z = 1.0f / (1.0f + expf(-((gi[1][tid] + bih2[HID + j]) + (gh[1][tid] + bhh2[HID + j]))));
    float nn = tanhf((gi[2][tid] + bih2[2 * HID + j]) + r * (gh[2][tid] + bhh2[2 * HID + j]));
    float out = (1.0f - z) * nn + z * hstate[HID + j];
    h2[j] = out;
    h2b[j] = f2bf(out);
  }
}

// ---------------- masked logits + Gumbel (512-rule block only) ----------------
// 32 blocks x 1 wave; block i handles rules [sym*512 + 16*i, +16).
__global__ void k_logits(const int* __restrict__ ctl,
                         const unsigned short* __restrict__ Waf,
                         const float* __restrict__ ba,
                         const unsigned short* __restrict__ h2b,
                         float* __restrict__ best_val, int* __restrict__ best_idx,
                         int t) {
  if (!ctl[0]) return;
  int lane = threadIdx.x & 31;
  size_t rt = (size_t)ctl[1] * 32 + blockIdx.x;   // Wa KT = 1024/32 = 32
  v8f acc = wave_matvec(Waf, rt * 32, 0, 32, h2b, lane);
  unsigned fk0, fk1;
  threefry2x32(0u, 42u, 0u, (unsigned)t, &fk0, &fk1);  // fold_in(key(42), t)
  __shared__ float sv[2];
  __shared__ int si[2];
  if ((lane & 15) == 0) {
    int half = lane >> 4;
    float bv = -1e30f; int bi = 0;
    #pragma unroll
    for (int r = 0; r < 8; ++r) {
      int row = (int)rt * 16 + half * 8 + r;
      float val = acc[r] + ba[row] + gumbel_for(fk0, fk1, row);
      if (val > bv) { bv = val; bi = row; }
    }
    sv[half] = bv; si[half] = bi;
  }
  __syncthreads();
  if (threadIdx.x == 0) {
    float bv = sv[0]; int bi = si[0];
    if (sv[1] > bv) { bv = sv[1]; bi = si[1]; }
    best_val[blockIdx.x] = bv;
    best_idx[blockIdx.x] = bi;
  }
}

// ---------------- sample / stack update / h commit / next-x gather ----------------
__global__ void k_update(int* __restrict__ ctl, int* __restrict__ stack,
                         const int* __restrict__ children,
                         const float* __restrict__ h1, const float* __restrict__ h2,
                         float* __restrict__ hstate, unsigned short* __restrict__ hstateb,
                         const float* __restrict__ emb, unsigned short* __restrict__ xvecb,
                         const float* __restrict__ best_val, const int* __restrict__ best_idx,
                         float* __restrict__ outF, int t) {
  __shared__ int s_exp;
  int tid = threadIdx.x;
  if (tid == 0) {
    int expand = ctl[0];
    s_exp = expand;
    int* actions = (int*)outF;
    if (expand) {
      float bv = best_val[0]; int bi = best_idx[0];
      for (int i = 1; i < 32; ++i)
        if (best_val[i] > bv) { bv = best_val[i]; bi = best_idx[i]; }
      int action = bi;
      int idx = ctl[5];
      int j1 = idx + 1; if (j1 > MAX_STACK - 1) j1 = MAX_STACK - 1;
      int ch0 = children[action * 2 + 0];
      int ch1 = children[action * 2 + 1];
      stack[idx * 2 + 0] = ch1; stack[idx * 2 + 1] = action;
      stack[j1 * 2 + 0]  = ch0; stack[j1 * 2 + 1]  = action;
      int np = idx + 2; if (np > MAX_STACK) np = MAX_STACK;
      ctl[4] = np;
      ctl[3] = action;
      actions[t] = action;
    } else {
      if (ctl[4] > 0) ctl[4] = ctl[4] - 1;
      actions[t] = PADV;
    }
    int ptr = ctl[4];
    int idx2 = ptr - 1; if (idx2 < 0) idx2 = 0;
    int sym = stack[idx2 * 2 + 0];
    int par = stack[idx2 * 2 + 1];
    ctl[5] = idx2;
    ctl[1] = sym;
    ctl[2] = par;
    ctl[0] = (ptr > 0 && sym < NUM_NT) ? 1 : 0;
  }
  __syncthreads();
  if (s_exp) {
    for (int i = tid; i < HID; i += blockDim.x) {
      float a = h1[i], b = h2[i];
      hstate[i] = a;
      hstate[HID + i] = b;
      hstateb[i] = f2bf(a);
      hstateb[HID + i] = f2bf(b);
    }
  }
  __syncthreads();
  int prev = ctl[3], par = ctl[2];
  for (int i = tid; i < EMBD; i += blockDim.x) {
    xvecb[i] = f2bf(emb[(size_t)(prev + 1) * EMBD + i]);
    xvecb[EMBD + i] = f2bf(emb[(size_t)(par + 1) * EMBD + i]);
  }
}

__global__ void k_final(const float* __restrict__ hstate, float* __restrict__ outF) {
  for (int i = threadIdx.x; i < 2 * HID; i += blockDim.x)
    outF[MAX_STEPS + i] = hstate[i];
}

extern "C" void kernel_launch(void* const* d_in, const int* in_sizes, int n_in,
                              void* d_out, int out_size, void* d_ws, size_t ws_size,
                              hipStream_t stream) {
  (void)in_sizes; (void)n_in; (void)out_size;
  const float* emb  = (const float*)d_in[0];
  const float* Wih1 = (const float*)d_in[1];
  const float* Whh1 = (const float*)d_in[2];
  const float* bih1 = (const float*)d_in[3];
  const float* bhh1 = (const float*)d_in[4];
  const float* Wih2 = (const float*)d_in[5];
  const float* Whh2 = (const float*)d_in[6];
  const float* bih2 = (const float*)d_in[7];
  const float* bhh2 = (const float*)d_in[8];
  const float* Wa   = (const float*)d_in[9];
  const float* ba   = (const float*)d_in[10];
  const float* h0   = (const float*)d_in[11];
  const int* children = (const int*)d_in[12];
  float* outF = (float*)d_out;

  char* ws = (char*)d_ws;
  size_t off = 0;
  auto take = [&](size_t bytes) -> void* {
    void* p = (void*)(ws + off);
    off += (bytes + 255) & ~(size_t)255;
    return p;
  };
  int*   ctl      = (int*)take(64);
  int*   stack    = (int*)take((size_t)MAX_STACK * 2 * sizeof(int));
  float* h1       = (float*)take((size_t)HID * sizeof(float));
  float* h2       = (float*)take((size_t)HID * sizeof(float));
  float* hstate   = (float*)take((size_t)2 * HID * sizeof(float));
  float* best_val = (float*)take(32 * sizeof(float));
  int*   best_idx = (int*)take(32 * sizeof(int));
  unsigned short* xvecb   = (unsigned short*)take((size_t)2 * EMBD * 2);
  unsigned short* h1b     = (unsigned short*)take((size_t)HID * 2);
  unsigned short* h2b     = (unsigned short*)take((size_t)HID * 2);
  unsigned short* hstateb = (unsigned short*)take((size_t)2 * HID * 2);
  unsigned short* Wih1f = (unsigned short*)take((size_t)3 * HID * 2 * EMBD * 2);
  unsigned short* Whh1f = (unsigned short*)take((size_t)3 * HID * HID * 2);
  unsigned short* Wih2f = (unsigned short*)take((size_t)3 * HID * HID * 2);
  unsigned short* Whh2f = (unsigned short*)take((size_t)3 * HID * HID * 2);
  unsigned short* Waf   = (unsigned short*)take((size_t)NUM_RULES * HID * 2);
  if (off > ws_size) return;  // workspace too small; nothing safe to do

  // One-time (per launch) bf16 fragment-major weight conversion — whole set
  // (~48 MB bf16) stays resident in the 192 MB L2 across all 1024 steps.
  k_conv<<<2048, 256, 0, stream>>>(Wih1, Wih1f, 3 * HID, 2 * EMBD);
  k_conv<<<2048, 256, 0, stream>>>(Whh1, Whh1f, 3 * HID, HID);
  k_conv<<<2048, 256, 0, stream>>>(Wih2, Wih2f, 3 * HID, HID);
  k_conv<<<2048, 256, 0, stream>>>(Whh2, Whh2f, 3 * HID, HID);
  k_conv<<<2048, 256, 0, stream>>>(Wa,   Waf,   NUM_RULES, HID);

  k_init<<<1, 256, 0, stream>>>(ctl, stack, h0, hstate, hstateb, emb, xvecb);

  for (int t = 0; t < MAX_STEPS; ++t) {
    k_gru1<<<64, 288, 0, stream>>>(ctl, Wih1f, Whh1f, bih1, bhh1,
                                   xvecb, hstate, hstateb, h1, h1b);
    k_gru2<<<64, 192, 0, stream>>>(ctl, Wih2f, Whh2f, bih2, bhh2,
                                   h1b, hstate, hstateb, h2, h2b);
    k_logits<<<32, 32, 0, stream>>>(ctl, Waf, ba, h2b, best_val, best_idx, t);
    k_update<<<1, 256, 0, stream>>>(ctl, stack, children, h1, h2, hstate, hstateb,
                                    emb, xvecb, best_val, best_idx, outF, t);
  }
  k_final<<<1, 256, 0, stream>>>(hstate, outF);
}
